// SGVAE_53060025974870
// MI455X (gfx1250) — matrix operand order repaired and checked by reading, not compiled, not run on hardware
//
#include <hip/hip_runtime.h>
#include <hip/hip_bf16.h>

typedef __attribute__((ext_vector_type(16))) _Float16 v16h;
typedef __attribute__((ext_vector_type(8)))  _Float16 v8h;
typedef __attribute__((ext_vector_type(8)))  float    v8f;
typedef __attribute__((ext_vector_type(4)))  float    v4f;

#define NNODES 8192
#define NEDGES 262144
#define FDIM   256
#define LAT    128

// ---------------------------------------------------------------- RNG helpers
__device__ __forceinline__ unsigned pcg_hash(unsigned v) {
  unsigned s = v * 747796405u + 2891336453u;
  unsigned w = ((s >> ((s >> 28u) + 4u)) ^ s) * 277803737u;
  return (w >> 22u) ^ w;
}

__device__ __forceinline__ float gaussian(unsigned idx) {
  unsigned a = pcg_hash(idx);
  unsigned b = pcg_hash(idx ^ 0x9e3779b9u);
  float u1 = ((float)a + 1.0f) * 2.3283064365386963e-10f;   // (0,1]
  float u2 = (float)b * 2.3283064365386963e-10f;            // [0,1)
  return sqrtf(-2.0f * __logf(u1)) * __cosf(6.28318530717958647f * u2);
}

// ---------------------------------------------------------------- aggregation
__global__ void count_kernel(const long long* __restrict__ ei,
                             float* __restrict__ cnt) {
  int e = blockIdx.x * blockDim.x + threadIdx.x;
  if (e < NEDGES) {
    int d = (int)ei[NEDGES + e];
    __hip_atomic_fetch_add(cnt + d, 1.0f, __ATOMIC_RELAXED, __HIP_MEMORY_SCOPE_AGENT);
  }
}

__global__ void invcnt_kernel(const float* __restrict__ cnt,
                              float* __restrict__ inv) {
  int i = blockIdx.x * blockDim.x + threadIdx.x;
  if (i < NNODES) inv[i] = 1.0f / fmaxf(cnt[i], 1.0f);
}

// one block per edge, 256 threads = 256 features
__global__ void agg_kernel(const long long* __restrict__ ei,
                           const float* __restrict__ x,
                           float* __restrict__ agg) {
  int e = blockIdx.x;
  int f = threadIdx.x;
  int s = (int)ei[e];
  int d = (int)ei[NEDGES + e];
  __hip_atomic_fetch_add(agg + (size_t)d * FDIM + f, x[(size_t)s * FDIM + f],
                         __ATOMIC_RELAXED, __HIP_MEMORY_SCOPE_AGENT);
}

// ---------------------------------------------------------------- weight packing
// Pack row-major f32 W[K][NO] into f16 in the exact wave32 B-tile layout:
// Wp[((kt*ntn + tn)*32 + lane)*16 + i]  holds  W[kt*32 + (lane&16?8:0) + i + (i>=8?8:0)][tn*16 + (lane&15)]
__global__ void pack_b_kernel(const float* __restrict__ W,
                              _Float16* __restrict__ Wp, int K, int NO) {
  int idx = blockIdx.x * blockDim.x + threadIdx.x;
  if (idx >= K * NO) return;
  int i    = idx & 15;
  int lane = (idx >> 4) & 31;
  int tile = idx >> 9;
  int ntn  = NO >> 4;
  int kt   = tile / ntn;
  int tn   = tile - kt * ntn;
  int n = (tn << 4) + (lane & 15);
  int k = (kt << 5) + ((lane & 16) ? 8 : 0) + i + ((i & 8) ? 8 : 0);
  Wp[idx] = (_Float16)W[(size_t)k * NO + n];
}

// ---------------------------------------------------------------- WMMA tile loaders
// A-tile 16x32 f16 from row-major f32 (ISA 7.12.2 layout), via two aligned
// float4 pairs -> global_load_b128 + packed cvt.
__device__ __forceinline__ v16h load_a16x32_f32(const float* __restrict__ A, int lda,
                                                int row, int k0, int lane, float scale) {
  const float* base = A + (size_t)row * lda + k0 + ((lane & 16) ? 8 : 0);
  v4f x0 = *(const v4f*)(base);
  v4f x1 = *(const v4f*)(base + 4);
  v4f x2 = *(const v4f*)(base + 16);
  v4f x3 = *(const v4f*)(base + 20);
  v16h a;
#pragma unroll
  for (int i = 0; i < 4; ++i) {
    a[i]      = (_Float16)(x0[i] * scale);
    a[4 + i]  = (_Float16)(x1[i] * scale);
    a[8 + i]  = (_Float16)(x2[i] * scale);
    a[12 + i] = (_Float16)(x3[i] * scale);
  }
  return a;
}

// B-tile from pre-packed f16: two contiguous 16B loads per lane.
__device__ __forceinline__ v16h load_bpacked(const _Float16* __restrict__ Wp,
                                             int ntn, int kt, int tn, int lane) {
  const _Float16* p = Wp + (((size_t)(kt * ntn + tn) * 32 + lane) << 4);
  v8h lo = *(const v8h*)(p);
  v8h hi = *(const v8h*)(p + 8);
  v16h b;
#pragma unroll
  for (int i = 0; i < 8; ++i) { b[i] = lo[i]; b[8 + i] = hi[i]; }
  return b;
}

// ---------------------------------------------------------------- fused SAGE linear
// out = (agg*invc) @ Wl + bias + xin @ Wr  [+ReLU]
// One wave computes a 16x64 strip (4 N-tiles): A loaded once per K-step,
// 8 v_wmma per K-step.
__global__ void sage_gemm_kernel(const float* __restrict__ agg,
                                 const float* __restrict__ invc,
                                 const float* __restrict__ xin,
                                 const _Float16* __restrict__ Wlp,
                                 const _Float16* __restrict__ Wrp,
                                 const float* __restrict__ bias,
                                 float* __restrict__ out,
                                 int K, int NO, int relu) {
  int lane = threadIdx.x & 31;
  int wave = blockIdx.x * (blockDim.x >> 5) + (threadIdx.x >> 5);
  int ngroups = NO >> 6;                      // groups of 4 tiles
  int tm = wave / ngroups;
  int tg = wave - tm * ngroups;
  int ntn = NO >> 4;
  int row = (tm << 4) + (lane & 15);
  float sc = invc[row];

  v8f acc[4] = {{}, {}, {}, {}};
  for (int kt = 0; kt < (K >> 5); ++kt) {
    v16h a1 = load_a16x32_f32(agg, K, row, kt << 5, lane, sc);
    v16h a2 = load_a16x32_f32(xin, K, row, kt << 5, lane, 1.0f);
#pragma unroll
    for (int j = 0; j < 4; ++j) {
      int tn = (tg << 2) + j;
      v16h b1 = load_bpacked(Wlp, ntn, kt, tn, lane);
      acc[j] = __builtin_amdgcn_wmma_f32_16x16x32_f16(false, a1, false, b1,
                                                      (short)0, acc[j], false, false);
      v16h b2 = load_bpacked(Wrp, ntn, kt, tn, lane);
      acc[j] = __builtin_amdgcn_wmma_f32_16x16x32_f16(false, a2, false, b2,
                                                      (short)0, acc[j], false, false);
    }
  }

  int mbase = (tm << 4) + ((lane & 16) ? 8 : 0);
#pragma unroll
  for (int j = 0; j < 4; ++j) {
    int col = (((tg << 2) + j) << 4) + (lane & 15);
    float bv = bias[col];
#pragma unroll
    for (int r = 0; r < 8; ++r) {
      float v = acc[j][r] + bv;
      if (relu) v = fmaxf(v, 0.0f);
      out[(size_t)(mbase + r) * NO + col] = v;
    }
  }
}

// ---------------------------------------------------------------- reparam + L2 norm
// one wave per node (LAT=128 -> 4 elements/lane)
__global__ void reparam_kernel(const float* __restrict__ mu,
                               const float* __restrict__ lv,
                               float* __restrict__ z,
                               _Float16* __restrict__ zn) {
  int lane = threadIdx.x & 31;
  int node = blockIdx.x * (blockDim.x >> 5) + (threadIdx.x >> 5);
  float zv[4];
  float ss = 0.0f;
#pragma unroll
  for (int j = 0; j < 4; ++j) {
    int idx = node * LAT + lane + j * 32;
    float eps = gaussian((unsigned)idx);
    float val = mu[idx] + eps * __expf(0.5f * lv[idx]);
    zv[j] = val;
    ss += val * val;
  }
#pragma unroll
  for (int o = 16; o > 0; o >>= 1) ss += __shfl_xor(ss, o, 32);
  float inv = 1.0f / (sqrtf(ss) + 1e-8f);
#pragma unroll
  for (int j = 0; j < 4; ++j) {
    int idx = node * LAT + lane + j * 32;
    z[idx]  = zv[j];
    zn[idx] = (_Float16)(zv[j] * inv);
  }
}

// ---------------------------------------------------------------- decoder GEMM
// recon = zn @ zn^T. One wave computes a 16x64 strip: A (16x128) held in
// registers, 16 v_wmma per wave, all loads 128-bit from L2-resident zn.
__global__ void decode_kernel(const _Float16* __restrict__ zn,
                              float* __restrict__ out) {
  int lane = threadIdx.x & 31;
  int wave = blockIdx.x * (blockDim.x >> 5) + (threadIdx.x >> 5);
  const int NG = NNODES >> 6;                 // 128 column groups of 64
  int tm = wave / NG;
  int tg = wave - tm * NG;

  // A strip: rows tm*16..tm*16+15, K = 0..127
  const _Float16* pa = zn + (size_t)((tm << 4) + (lane & 15)) * LAT
                          + ((lane & 16) ? 8 : 0);
  v16h a[4];
#pragma unroll
  for (int t = 0; t < 4; ++t) {
    v8h lo = *(const v8h*)(pa + t * 32);
    v8h hi = *(const v8h*)(pa + t * 32 + 16);
#pragma unroll
    for (int i = 0; i < 8; ++i) { a[t][i] = lo[i]; a[t][8 + i] = hi[i]; }
  }

  int mbase = (tm << 4) + ((lane & 16) ? 8 : 0);
#pragma unroll
  for (int j = 0; j < 4; ++j) {
    const _Float16* pb = zn + (size_t)((tg << 6) + (j << 4) + (lane & 15)) * LAT
                            + ((lane & 16) ? 8 : 0);
    v8f acc = {};
#pragma unroll
    for (int t = 0; t < 4; ++t) {
      v8h lo = *(const v8h*)(pb + t * 32);
      v8h hi = *(const v8h*)(pb + t * 32 + 16);
      v16h b;
#pragma unroll
      for (int i = 0; i < 8; ++i) { b[i] = lo[i]; b[8 + i] = hi[i]; }
      acc = __builtin_amdgcn_wmma_f32_16x16x32_f16(false, a[t], false, b,
                                                   (short)0, acc, false, false);
    }
    int col = (tg << 6) + (j << 4) + (lane & 15);
#pragma unroll
    for (int r = 0; r < 8; ++r)
      out[(size_t)(mbase + r) * NNODES + col] = acc[r];
  }
}

// ---------------------------------------------------------------- launch
extern "C" void kernel_launch(void* const* d_in, const int* in_sizes, int n_in,
                              void* d_out, int out_size, void* d_ws, size_t ws_size,
                              hipStream_t stream) {
  (void)in_sizes; (void)n_in; (void)out_size; (void)ws_size;
  const float*     x   = (const float*)d_in[0];
  const long long* ei  = (const long long*)d_in[1];
  const float*     Wl1 = (const float*)d_in[2];
  const float*     bl1 = (const float*)d_in[3];
  const float*     Wr1 = (const float*)d_in[4];
  const float*     Wl2 = (const float*)d_in[5];
  const float*     bl2 = (const float*)d_in[6];
  const float*     Wr2 = (const float*)d_in[7];
  const float*     Wl3 = (const float*)d_in[8];
  const float*     bl3 = (const float*)d_in[9];
  const float*     Wr3 = (const float*)d_in[10];

  // output regions (return order: recon, mu, logvar, z)
  float* recon = (float*)d_out;
  float* mu    = recon + (size_t)NNODES * NNODES;
  float* lvp   = mu    + (size_t)NNODES * LAT;
  float* zout  = lvp   + (size_t)NNODES * LAT;

  // workspace layout
  char*  ws   = (char*)d_ws;
  size_t o = 0;
  float* agg  = (float*)(ws + o);  o += (size_t)NNODES * FDIM * 4;
  float* cnt  = (float*)(ws + o);  o += (size_t)NNODES * 4;
  float* invc = (float*)(ws + o);  o += (size_t)NNODES * 4;
  float* h    = (float*)(ws + o);  o += (size_t)NNODES * FDIM * 4;
  _Float16* zn   = (_Float16*)(ws + o);  o += (size_t)NNODES * LAT * 2;
  _Float16* Wl1p = (_Float16*)(ws + o);  o += (size_t)FDIM * FDIM * 2;
  _Float16* Wr1p = (_Float16*)(ws + o);  o += (size_t)FDIM * FDIM * 2;
  _Float16* Wl2p = (_Float16*)(ws + o);  o += (size_t)FDIM * LAT * 2;
  _Float16* Wr2p = (_Float16*)(ws + o);  o += (size_t)FDIM * LAT * 2;
  _Float16* Wl3p = (_Float16*)(ws + o);  o += (size_t)FDIM * LAT * 2;
  _Float16* Wr3p = (_Float16*)(ws + o);  o += (size_t)FDIM * LAT * 2;

  // pack weights into WMMA B-tile layout (f16)
  pack_b_kernel<<<(FDIM * FDIM) / 256, 256, 0, stream>>>(Wl1, Wl1p, FDIM, FDIM);
  pack_b_kernel<<<(FDIM * FDIM) / 256, 256, 0, stream>>>(Wr1, Wr1p, FDIM, FDIM);
  pack_b_kernel<<<(FDIM * LAT) / 256, 256, 0, stream>>>(Wl2, Wl2p, FDIM, LAT);
  pack_b_kernel<<<(FDIM * LAT) / 256, 256, 0, stream>>>(Wr2, Wr2p, FDIM, LAT);
  pack_b_kernel<<<(FDIM * LAT) / 256, 256, 0, stream>>>(Wl3, Wl3p, FDIM, LAT);
  pack_b_kernel<<<(FDIM * LAT) / 256, 256, 0, stream>>>(Wr3, Wr3p, FDIM, LAT);

  // degrees (shared by all three convs)
  hipMemsetAsync(cnt, 0, (size_t)NNODES * 4, stream);
  count_kernel<<<NEDGES / 256, 256, 0, stream>>>(ei, cnt);
  invcnt_kernel<<<NNODES / 256, 256, 0, stream>>>(cnt, invc);

  // conv1: h = relu(mean(x) @ Wl1 + bl1 + x @ Wr1)
  hipMemsetAsync(agg, 0, (size_t)NNODES * FDIM * 4, stream);
  agg_kernel<<<NEDGES, FDIM, 0, stream>>>(ei, x, agg);
  {
    int waves = (NNODES / 16) * (FDIM / 64);       // 2048
    sage_gemm_kernel<<<waves / 4, 128, 0, stream>>>(agg, invc, x, Wl1p, Wr1p, bl1,
                                                    h, FDIM, FDIM, 1);
  }

  // aggregate h once, reuse for conv2 + conv3
  hipMemsetAsync(agg, 0, (size_t)NNODES * FDIM * 4, stream);
  agg_kernel<<<NEDGES, FDIM, 0, stream>>>(ei, h, agg);
  {
    int waves = (NNODES / 16) * (LAT / 64);        // 1024
    sage_gemm_kernel<<<waves / 4, 128, 0, stream>>>(agg, invc, h, Wl2p, Wr2p, bl2,
                                                    mu, FDIM, LAT, 0);
    sage_gemm_kernel<<<waves / 4, 128, 0, stream>>>(agg, invc, h, Wl3p, Wr3p, bl3,
                                                    lvp, FDIM, LAT, 0);
  }

  // reparameterize + L2 normalize (one wave per node)
  reparam_kernel<<<NNODES / 4, 128, 0, stream>>>(mu, lvp, zout, zn);

  // decoder: recon = zn @ zn^T  (65536 waves of 16x64 strips)
  {
    int waves = (NNODES / 16) * (NNODES / 64);     // 65536
    decode_kernel<<<waves / 8, 256, 0, stream>>>(zn, recon);
  }
}
